// EdgeUpdateNetBackbone_30932354466269
// MI455X (gfx1250) — matrix operand order repaired and checked by reading, not compiled, not run on hardware
//
#include <hip/hip_runtime.h>
#include <hip/hip_bf16.h>

typedef __attribute__((ext_vector_type(16))) _Float16 v16h;
typedef __attribute__((ext_vector_type(8)))  _Float16 v8h;
typedef __attribute__((ext_vector_type(8)))  float    v8f;

#define CCH 64   // hidden channels
#define NGS 128  // num gaussians
#define MAXNK 8  // max K/32 steps (K<=256)
#define RT 4     // row tiles (accumulators) per wave

__device__ __forceinline__ float ssp_f(float x) {
    // shifted softplus: softplus(x) - log(2), numerically stable
    float ax = fabsf(x);
    return fmaxf(x, 0.0f) + __logf(1.0f + __expf(-ax)) - 0.6931471805599453f;
}

// ---------------------------------------------------------------------------
__global__ void emb_gather_kernel(const int* __restrict__ z,
                                  const float* __restrict__ emb,
                                  float* __restrict__ h, int total) {
    int i = blockIdx.x * 256 + threadIdx.x;
    if (i < total) {
        int n = i >> 6, c = i & 63;
        h[i] = emb[z[n] * CCH + c];
    }
}

// edge_attr[e,g] = envelope(d/cutoff) * exp(coeff*(ds-off_g)^2)
__global__ void rbf_kernel(const float* __restrict__ pos,
                           const int* __restrict__ rowi,
                           const int* __restrict__ coli,
                           float* __restrict__ attr, int E) {
    int e = blockIdx.x;
    int g = threadIdx.x;
    if (e >= E) return;
    int r = rowi[e], c = coli[e];
    float dx = pos[r * 3 + 0] - pos[c * 3 + 0];
    float dy = pos[r * 3 + 1] - pos[c * 3 + 1];
    float dz = pos[r * 3 + 2] - pos[c * 3 + 2];
    float d  = sqrtf(dx * dx + dy * dy + dz * dz);
    float ds = d * 0.1f;
    float env = 0.0f;
    if (ds < 1.0f) {
        float ds5 = ds * ds;
        ds5 = ds5 * ds5 * ds;
        env = 1.0f + (-21.0f) * ds5 + 35.0f * ds5 * ds + (-15.0f) * ds5 * ds * ds;
    }
    const float delta = 1.0f / 127.0f;
    const float coeff = -0.5f / (delta * delta);
    float t = ds - (float)g * delta;
    attr[(size_t)e * NGS + g] = env * __expf(coeff * t * t);
}

// ---------------------------------------------------------------------------
// Fused WMMA GEMM:  Out[M,Nout] = act(A[M,K] @ W[K,Nout] + bias) (+residual)
//   mode 0: A dense row-major [M,K]
//   mode 1: A = concat(h[row[m]], h[col[m]], attr[m][0:attr_ld])
//
// 256 threads = 8 waves. tiles_n = Nout/16 col tiles; row_waves = 8/tiles_n;
// each wave owns RT=4 row tiles -> rpb = row_waves*64 rows per block.
// B fragments preloaded into registers (contiguous k = 16*(lane/16) + i, same as A), reused
// by 4 wmmas per K-step. A tile double-buffered in LDS (f16), staged in
// 8-wide chunks (2x float4 load -> packed cvt -> ds_store_b128); exactly one
// barrier per K-step.
// ---------------------------------------------------------------------------
__global__ void wmma_gemm_kernel(const float* __restrict__ A,
                                 const float* __restrict__ hsrc,
                                 const float* __restrict__ attr,
                                 const int* __restrict__ rowi,
                                 const int* __restrict__ coli,
                                 int attr_ld,
                                 const float* __restrict__ W,
                                 const float* __restrict__ bias,
                                 const float* __restrict__ residual,
                                 float* __restrict__ Out,
                                 int M, int K, int Nout,
                                 int mode, int act) {
    extern __shared__ _Float16 smem[];

    const int tid  = threadIdx.x;
    const int lane = tid & 31;
    const int wave = tid >> 5;
    const int tiles_n = Nout >> 4;            // 2, 4, or 8
    const int row_waves = 8 / tiles_n;
    const int rpb = row_waves * (RT * 16);    // rows per block
    const int wr = wave / tiles_n;
    const int wc = wave - wr * tiles_n;
    const int blk_row0 = blockIdx.x * rpb;
    const int n0 = wc * 16;
    const int nk = K >> 5;

    _Float16* sbuf0 = smem;                   // [rpb][32] f16 A tile, buffer 0
    _Float16* sbuf1 = smem + (size_t)rpb * 32;
    int* sIdx = (int*)(smem + (size_t)2 * rpb * 32);  // [2][rpb] gather indices

    // Hoist gather indices (K-invariant) into LDS
    if (mode == 1) {
        for (int r = tid; r < rpb; r += 256) {
            int gr = blk_row0 + r;
            sIdx[r]       = (gr < M) ? rowi[gr] : 0;
            sIdx[rpb + r] = (gr < M) ? coli[gr] : 0;
        }
    }

    // Preload B fragments into registers: bfr[s][h] = W[(s*32+bko+h)*Nout + bn]
    v16h bfr[MAXNK];
    const int bn  = n0 + (lane & 15);
    const int bko = (lane >> 4) * 16;
#pragma unroll
    for (int s = 0; s < MAXNK; s++) {
        if (s < nk) {
            const float* wp = W + (size_t)(s * 32 + bko) * Nout + bn;
#pragma unroll
            for (int hI = 0; hI < 16; hI++)
                bfr[s][hI] = (_Float16)wp[(size_t)hI * Nout];
        }
    }

    // A-tile stage: rpb rows x 32 K, 8-halves per chunk, vectorized
    auto stage = [&](int k0, _Float16* dst) {
        const int nchunk = rpb * 4;           // 4 chunks of 8 per row
        for (int ch = tid; ch < nchunk; ch += 256) {
            int r  = ch >> 2;
            int kk = (ch & 3) * 8;
            int gr = blk_row0 + r;
            int k  = k0 + kk;
            float4 f0 = make_float4(0.f, 0.f, 0.f, 0.f);
            float4 f1 = f0;
            if (gr < M) {
                const float* src;
                if (mode == 1) {
                    if (k < CCH)          src = hsrc + (size_t)sIdx[r] * CCH + k;
                    else if (k < 2 * CCH) src = hsrc + (size_t)sIdx[rpb + r] * CCH + (k - CCH);
                    else                  src = attr + (size_t)gr * attr_ld + (k - 2 * CCH);
                } else {
                    src = A + (size_t)gr * K + k;
                }
                f0 = *(const float4*)src;
                f1 = *(const float4*)(src + 4);
            }
            v8h hv;
            hv[0] = (_Float16)f0.x; hv[1] = (_Float16)f0.y;
            hv[2] = (_Float16)f0.z; hv[3] = (_Float16)f0.w;
            hv[4] = (_Float16)f1.x; hv[5] = (_Float16)f1.y;
            hv[6] = (_Float16)f1.z; hv[7] = (_Float16)f1.w;
            *(v8h*)(dst + (size_t)r * 32 + kk) = hv;
        }
    };

    __syncthreads();            // sIdx visible before first stage
    stage(0, sbuf0);

    v8f acc[RT];
#pragma unroll
    for (int t = 0; t < RT; t++) acc[t] = (v8f){};

    const int koff = (lane >> 4) * 16;
    int arow[RT];
#pragma unroll
    for (int t = 0; t < RT; t++) arow[t] = (wr * RT + t) * 16 + (lane & 15);

#pragma unroll
    for (int s = 0; s < MAXNK; s++) {
        if (s < nk) {
            __syncthreads();    // cur buffer staged; prev reads retired
            _Float16* cur = (s & 1) ? sbuf1 : sbuf0;
            _Float16* nxt = (s & 1) ? sbuf0 : sbuf1;
            if (s + 1 < nk) stage((s + 1) * 32, nxt);

#pragma unroll
            for (int t = 0; t < RT; t++) {
                v16h af;
                v8h lo = *(const v8h*)(cur + arow[t] * 32 + koff);
                v8h hi = *(const v8h*)(cur + arow[t] * 32 + koff + 8);
#pragma unroll
                for (int q = 0; q < 8; q++) { af[q] = lo[q]; af[q + 8] = hi[q]; }
                acc[t] = __builtin_amdgcn_wmma_f32_16x16x32_f16(
                    false, af, false, bfr[s], (short)0, acc[t], false, false);
            }
        }
    }

    // Store: n = lane&15, m = v + 8*(lane>=16)  (ISA C/D layout)
    const int n   = n0 + (lane & 15);
    const int mhi = (lane >> 4) * 8;
    const float bv = bias[n];
#pragma unroll
    for (int t = 0; t < RT; t++) {
        int rbase = blk_row0 + (wr * RT + t) * 16 + mhi;
#pragma unroll
        for (int v = 0; v < 8; v++) {
            int mrow = rbase + v;
            if (mrow < M) {
                float x = acc[t][v] + bv;
                if (act) x = ssp_f(x);
                if (residual) x += residual[(size_t)mrow * Nout + n];
                Out[(size_t)mrow * Nout + n] = x;
            }
        }
    }
}

// ---------------------------------------------------------------------------
__global__ void zero_kernel(float* __restrict__ p, int total) {
    int i = blockIdx.x * 256 + threadIdx.x;
    if (i < total) p[i] = 0.0f;
}

// m[col[e], c] += hh[row[e], c] * eg[e, c]
__global__ void scatter_kernel(const float* __restrict__ hh,
                               const float* __restrict__ eg,
                               const int* __restrict__ rowi,
                               const int* __restrict__ coli,
                               float* __restrict__ m, int E) {
    long long tid = (long long)blockIdx.x * 256 + threadIdx.x;
    if (tid < (long long)E * CCH) {
        int e = (int)(tid >> 6);
        int c = (int)(tid & 63);
        float v = hh[(size_t)rowi[e] * CCH + c] * eg[tid];
        atomicAdd(&m[(size_t)coli[e] * CCH + c], v);
    }
}

// out[n] = ob2 + sum_c t4[n,c]*ow2[c]
__global__ void final_dot_kernel(const float* __restrict__ t4,
                                 const float* __restrict__ ow2,
                                 const float* __restrict__ ob2,
                                 float* __restrict__ out, int N) {
    int n = blockIdx.x * 256 + threadIdx.x;
    if (n < N) {
        float s = ob2[0];
#pragma unroll
        for (int c = 0; c < 32; c++) s += t4[(size_t)n * 32 + c] * ow2[c];
        out[n] = s;
    }
}

// ---------------------------------------------------------------------------
static void launch_gemm(const float* A, const float* h, const float* attr,
                        const int* rowi, const int* coli, int attr_ld,
                        const float* W, const float* bias, const float* res,
                        float* Out, int M, int K, int Nout, int mode, int act,
                        hipStream_t s) {
    int tiles_n = Nout / 16;
    int rpb = (8 / tiles_n) * (RT * 16);
    int grid = (M + rpb - 1) / rpb;
    size_t shmem = (size_t)2 * rpb * 32 * sizeof(_Float16)   // double A buffer
                 + (size_t)2 * rpb * sizeof(int);            // gather indices
    wmma_gemm_kernel<<<dim3(grid), dim3(256), shmem, s>>>(
        A, h, attr, rowi, coli, attr_ld, W, bias, res, Out, M, K, Nout, mode, act);
}

extern "C" void kernel_launch(void* const* d_in, const int* in_sizes, int n_in,
                              void* d_out, int out_size, void* d_ws, size_t ws_size,
                              hipStream_t stream) {
    const int N = in_sizes[0];
    const int E = in_sizes[2] / 2;

    const int*   z    = (const int*)d_in[0];
    const float* pos  = (const float*)d_in[1];
    const int*   ei   = (const int*)d_in[2];
    const int*   rowi = ei;
    const int*   coli = ei + E;

    // Params flattened pytree-style (dict keys sorted):
    // edge[i]: b1,b2,w1,w2 -> 3..14 ; emb -> 15
    // inter[i]: fc1_b,fc1_w,m1b1,m1b2,m1w1,m1w2,mb1,mb2,mw1,mw2 -> 16+10i
    // ob1=46, ob2=47, ow1=48, ow2=49
    const float* e_b1[3] = {(const float*)d_in[3],  (const float*)d_in[7],  (const float*)d_in[11]};
    const float* e_b2[3] = {(const float*)d_in[4],  (const float*)d_in[8],  (const float*)d_in[12]};
    const float* e_w1[3] = {(const float*)d_in[5],  (const float*)d_in[9],  (const float*)d_in[13]};
    const float* e_w2[3] = {(const float*)d_in[6],  (const float*)d_in[10], (const float*)d_in[14]};
    const float* emb = (const float*)d_in[15];
    const float *fc1_b[3], *fc1_w[3], *m1b1[3], *m1b2[3], *m1w1[3], *m1w2[3],
                *mb1[3], *mb2[3], *mw1[3], *mw2[3];
    for (int i = 0; i < 3; i++) {
        int b = 16 + 10 * i;
        fc1_b[i] = (const float*)d_in[b + 0];
        fc1_w[i] = (const float*)d_in[b + 1];
        m1b1[i]  = (const float*)d_in[b + 2];
        m1b2[i]  = (const float*)d_in[b + 3];
        m1w1[i]  = (const float*)d_in[b + 4];
        m1w2[i]  = (const float*)d_in[b + 5];
        mb1[i]   = (const float*)d_in[b + 6];
        mb2[i]   = (const float*)d_in[b + 7];
        mw1[i]   = (const float*)d_in[b + 8];
        mw2[i]   = (const float*)d_in[b + 9];
    }
    const float* ob1 = (const float*)d_in[46];
    const float* ob2 = (const float*)d_in[47];
    const float* ow1 = (const float*)d_in[48];
    const float* ow2 = (const float*)d_in[49];

    // Workspace layout (floats)
    float* ws  = (float*)d_ws;
    float* h   = ws;                      ws += (size_t)N * CCH;
    float* hh  = ws;                      ws += (size_t)N * CCH;
    float* m   = ws;                      ws += (size_t)N * CCH;
    float* t3  = ws;                      ws += (size_t)N * CCH;
    float* t4  = ws;                      ws += (size_t)N * 32;
    float* attr = ws;                     ws += (size_t)E * NGS;   // edge features
    float* buf1 = ws;                                              // E*128 ping
    float* t2   = buf1;                   // lower E*64 half (after buf1 consumed)
    float* eg   = buf1 + (size_t)E * CCH; // upper E*64 half

    emb_gather_kernel<<<(N * CCH + 255) / 256, 256, 0, stream>>>(z, emb, h, N * CCH);
    rbf_kernel<<<E, NGS, 0, stream>>>(pos, rowi, coli, attr, E);

    for (int i = 0; i < 3; i++) {
        const int in_f = (i == 0) ? (NGS + 2 * CCH) : (3 * CCH);   // 256 or 192
        const int a_ld = (i == 0) ? NGS : CCH;
        // EdgeUpdate: buf1 = ssp(cat @ w1 + b1)   [E, 128]
        launch_gemm(nullptr, h, attr, rowi, coli, a_ld,
                    e_w1[i], e_b1[i], nullptr, buf1, E, in_f, 2 * CCH, 1, 1, stream);
        // attr = buf1 @ w2 + b2                    [E, 64]
        launch_gemm(buf1, nullptr, nullptr, nullptr, nullptr, 0,
                    e_w2[i], e_b2[i], nullptr, attr, E, 2 * CCH, CCH, 0, 0, stream);
        // hh = h @ fc1_w + fc1_b                   [N, 64]
        launch_gemm(h, nullptr, nullptr, nullptr, nullptr, 0,
                    fc1_w[i], fc1_b[i], nullptr, hh, N, CCH, CCH, 0, 0, stream);
        // t2 = ssp(attr @ mw1 + mb1); eg = ssp(t2 @ mw2 + mb2)   [E, 64]
        launch_gemm(attr, nullptr, nullptr, nullptr, nullptr, 0,
                    mw1[i], mb1[i], nullptr, t2, E, CCH, CCH, 0, 1, stream);
        launch_gemm(t2, nullptr, nullptr, nullptr, nullptr, 0,
                    mw2[i], mb2[i], nullptr, eg, E, CCH, CCH, 0, 1, stream);
        // m = segment_sum(hh[row] * eg, col)
        zero_kernel<<<(N * CCH + 255) / 256, 256, 0, stream>>>(m, N * CCH);
        long long tot = (long long)E * CCH;
        scatter_kernel<<<(int)((tot + 255) / 256), 256, 0, stream>>>(hh, eg, rowi, coli, m, E);
        // t3 = ssp(m @ m1w1 + m1b1); h = h + (t3 @ m1w2 + m1b2)
        launch_gemm(m, nullptr, nullptr, nullptr, nullptr, 0,
                    m1w1[i], m1b1[i], nullptr, t3, N, CCH, CCH, 0, 1, stream);
        launch_gemm(t3, nullptr, nullptr, nullptr, nullptr, 0,
                    m1w2[i], m1b2[i], h, h, N, CCH, CCH, 0, 0, stream);
    }

    launch_gemm(h, nullptr, nullptr, nullptr, nullptr, 0,
                ow1, ob1, nullptr, t4, N, CCH, 32, 0, 1, stream);
    final_dot_kernel<<<(N + 255) / 256, 256, 0, stream>>>(t4, ow2, ob2, (float*)d_out, N);
    (void)n_in; (void)out_size; (void)ws_size;
}
